// MultiHeadAttention_79508434583981
// MI455X (gfx1250) — compile-verified, matrix operand order checked by reading, not run on hardware
//
#include <hip/hip_runtime.h>
#include <hip/hip_bf16.h>

// ---------------------------------------------------------------------------
// MultiHeadAttention for MI455X (gfx1250, wave32, WMMA bf16 16x16x32 + TDM).
// B=4, S=2048, D_MODEL=1024, H=16, HEAD_DIM=64.
// ---------------------------------------------------------------------------

#define B_SZ   4
#define S_LEN  2048
#define DM     1024
#define NH     16
#define HD     64
#define QKV_N  3072            // 3*DM
#define MROWS  (B_SZ * S_LEN)  // 8192

typedef __attribute__((ext_vector_type(16))) __bf16 v16bf;
typedef __attribute__((ext_vector_type(8)))  float  v8f;
typedef __attribute__((ext_vector_type(4)))  unsigned int v4u;
typedef __attribute__((ext_vector_type(8)))  int v8i;
typedef __attribute__((ext_vector_type(4)))  int v4i;

struct alignas(16) U4 { unsigned int x, y, z, w; };

union BF16Frag {
    v16bf          v;
    unsigned short h[16];
    U4             q[2];
};

__device__ __forceinline__ unsigned short f32_to_bf16(float f) {
    unsigned int u = __float_as_uint(f);
    unsigned int r = (u + 0x7FFFu + ((u >> 16) & 1u)) >> 16;  // RNE
    return (unsigned short)r;
}

// ---------------------------------------------------------------------------
// Tensor Data Mover: issue a 2-D tile load (rows x klen bf16, row stride
// `stride_elems`) from global into LDS. Descriptor packing per CDNA5 ISA §8.
// Tracked with TENSORcnt; wait with __builtin_amdgcn_s_wait_tensorcnt(0).
// This toolchain exposes the 6-arg builtin: (v4u, v8i, v4i, v4i, v8i, cpol).
// ---------------------------------------------------------------------------
__device__ __forceinline__ void tdm_load_tile(unsigned int lds_addr,
                                              const unsigned short* gptr,
                                              unsigned int klen,
                                              unsigned int rows,
                                              unsigned int stride_elems,
                                              unsigned int tensor_d0) {
    unsigned long long ga = (unsigned long long)(size_t)gptr;

    v4u g0;
    g0[0] = 1u;                                        // count=1, user D#
    g0[1] = lds_addr;                                  // LDS byte address
    g0[2] = (unsigned int)(ga & 0xFFFFFFFFu);          // global_addr[31:0]
    g0[3] = (unsigned int)((ga >> 32) & 0x01FFFFFFu)   // global_addr[56:32]
          | 0x80000000u;                               // type=2 ("image")

    const unsigned int td0 = tensor_d0;                // tensor_dim0 (>= tile reach)
    const unsigned int td1 = 0x00FFFFFFu;              // tensor_dim1 (large, no OOB)
    v8i g1;
    g1[0] = (int)(1u << 16);                           // data_size=1 -> 2 bytes
    g1[1] = (int)((td0 & 0xFFFFu) << 16);              // tensor_dim0[15:0]
    g1[2] = (int)(((td0 >> 16) & 0xFFFFu) | ((td1 & 0xFFFFu) << 16));
    g1[3] = (int)(((td1 >> 16) & 0xFFFFu) | (klen << 16));   // tile_dim0
    g1[4] = (int)(rows & 0xFFFFu);                     // tile_dim1 (tile_dim2=0)
    g1[5] = (int)stride_elems;                         // tensor_dim0_stride[31:0]
    g1[6] = 0;                                         // stride hi / dim1_stride lo
    g1[7] = 0;

    v4i g2 = {0, 0, 0, 0};
    v4i g3 = {0, 0, 0, 0};
    v8i g4 = {0, 0, 0, 0, 0, 0, 0, 0};
    __builtin_amdgcn_tensor_load_to_lds(g0, g1, g2, g3, g4, 0);
}

// ---------------------------------------------------------------------------
// fp32 -> bf16 elementwise conversion
// ---------------------------------------------------------------------------
__global__ void f2bf_kernel(const float* __restrict__ in,
                            unsigned short* __restrict__ out, int n) {
    int i = blockIdx.x * blockDim.x + threadIdx.x;
    if (i < n) out[i] = f32_to_bf16(in[i]);
}

// ---------------------------------------------------------------------------
// GEMM: C[m,n] = sum_k A[m,k] * W[n,k] + bias[n]
//   A: [M x K] bf16 row major, W: [N x K] bf16 (torch Linear layout).
// Block = 8 waves = 128 rows x 64 cols. The shared 64x32 B strip is staged
// into LDS by the TDM (double buffered, wave 0 issues, TENSORcnt + barriers
// sequence it); each wave then runs 4 v_wmma_f32_16x16x32_bf16 per K-step
// with A from global (b128) and B from LDS (ds_load_b128).
// Grid: (N/64, M/128), block 256.
// ---------------------------------------------------------------------------
__global__ __launch_bounds__(256)
void gemm_bf16_kernel(const unsigned short* __restrict__ A,
                      const unsigned short* __restrict__ W,
                      const float* __restrict__ bias,
                      unsigned short* __restrict__ Cb,
                      float* __restrict__ Cf,
                      int N, int K) {
    __shared__ alignas(16) unsigned short btile[2][64 * 32];

    const int wave = threadIdx.x >> 5;
    const int lane = threadIdx.x & 31;
    const int lrow = lane & 15;   // row (A) / col (B) index within 16-tile
    const int lhi  = lane >> 4;   // K-half selector per ISA 16-bit layouts

    const int m0 = blockIdx.y * 128 + wave * 16;
    const int n0 = blockIdx.x * 64;

    v8f acc[4] = {};
    const unsigned short* arow = A + (size_t)(m0 + lrow) * K;
    const unsigned short* wbase = W + (size_t)n0 * K;

    if (threadIdx.x < 32) {
        tdm_load_tile((unsigned int)(size_t)&btile[0][0], wbase,
                      32u, 64u, (unsigned int)K, (unsigned int)K);
    }

    for (int k = 0; k < K; k += 32) {
        const int buf = (k >> 5) & 1;
        if (threadIdx.x < 32) __builtin_amdgcn_s_wait_tensorcnt(0);
        __syncthreads();                       // btile[buf] ready for all waves
        if ((k + 32 < K) && threadIdx.x < 32) {
            tdm_load_tile((unsigned int)(size_t)&btile[buf ^ 1][0],
                          wbase + (k + 32),
                          32u, 64u, (unsigned int)K, (unsigned int)K);
        }

        BF16Frag a;
        a.q[0] = *(const U4*)(arow + k +      lhi * 8);
        a.q[1] = *(const U4*)(arow + k + 16 + lhi * 8);
        const unsigned short* bsm = btile[buf];
#pragma unroll
        for (int t = 0; t < 4; ++t) {
            BF16Frag bfr;
            bfr.q[0] = *(const U4*)(bsm + (t * 16 + lrow) * 32 +      lhi * 8);
            bfr.q[1] = *(const U4*)(bsm + (t * 16 + lrow) * 32 + 16 + lhi * 8);
            acc[t] = __builtin_amdgcn_wmma_f32_16x16x32_bf16(
                false, a.v, false, bfr.v, (short)0, acc[t], false, false);
        }
        __syncthreads();                       // all reads of btile[buf] done
    }

#pragma unroll
    for (int t = 0; t < 4; ++t) {
        const int n  = n0 + t * 16 + lrow;
        const float bv = bias[n];
#pragma unroll
        for (int v = 0; v < 8; ++v) {
            const int m  = m0 + v + 8 * lhi;
            const float val = acc[t][v] + bv;
            if (Cb) Cb[(size_t)m * N + n] = f32_to_bf16(val);
            else    Cf[(size_t)m * N + n] = val;
        }
    }
}

// ---------------------------------------------------------------------------
// Flash attention: one wave per 16-query tile of one (b,h); 4 waves/block
// share (b,h) so the 32-key x 64-dim V block is TDM-staged into LDS once per
// block (double buffered). Per 32-key block: 4 WMMA for S = Q K^T, online
// softmax (cross-lane shfl reductions), P reshaped via LDS into A-frag
// layout, 4 WMMA for O += P V (V B-frags read from LDS).
// qkv row layout (3072 bf16): [h][q(64) | k(64) | v(64)].
// Grid: (S/16/4, B*NH), block 128.
// ---------------------------------------------------------------------------
__global__ __launch_bounds__(128)
void attn_kernel(const unsigned short* __restrict__ qkv,
                 unsigned short* __restrict__ vout) {
    __shared__ alignas(16) unsigned short plds[4][16 * 32];
    __shared__ alignas(16) unsigned short vtile[2][32 * 64];

    const int wave = threadIdx.x >> 5;
    const int lane = threadIdx.x & 31;
    const int lrow = lane & 15;
    const int lhi  = lane >> 4;

    const int bh = blockIdx.y;        // 0..63
    const int b  = bh >> 4;
    const int h  = bh & 15;
    const int q0 = (blockIdx.x * 4 + wave) * 16;

    const size_t rs = QKV_N;
    const unsigned short* base  = qkv + (size_t)b * S_LEN * rs + (size_t)h * 192;
    const unsigned short* vbase = base + 128;

    // Q A-frags (16 x 64 as 2 x (16x32))
    BF16Frag qa[2];
    {
        const unsigned short* qrow = base + (size_t)(q0 + lrow) * rs;
        qa[0].q[0] = *(const U4*)(qrow +  0 + lhi * 8);
        qa[0].q[1] = *(const U4*)(qrow + 16 + lhi * 8);
        qa[1].q[0] = *(const U4*)(qrow + 32 + lhi * 8);
        qa[1].q[1] = *(const U4*)(qrow + 48 + lhi * 8);
    }

    float m_i[8], l_i[8];
    v8f   o[4] = {};
#pragma unroll
    for (int v = 0; v < 8; ++v) { m_i[v] = -1e30f; l_i[v] = 0.0f; }
    const float scale = 0.125f;   // 1/sqrt(64)

    // prefetch first V block (32 keys x 64 dims, row stride 3072)
    if (threadIdx.x < 32) {
        tdm_load_tile((unsigned int)(size_t)&vtile[0][0], vbase,
                      64u, 32u, (unsigned int)rs, 64u);
    }

    for (int kb = 0; kb < S_LEN; kb += 32) {
        const int buf = (kb >> 5) & 1;

        // ---- scores: two 16x16 tiles over this 32-key block ----
        v8f s[2] = {};
#pragma unroll
        for (int t = 0; t < 2; ++t) {
            const unsigned short* krow =
                base + (size_t)(kb + t * 16 + lrow) * rs + 64;
            BF16Frag b0, b1;
            b0.q[0] = *(const U4*)(krow +  0 + lhi * 8);
            b0.q[1] = *(const U4*)(krow + 16 + lhi * 8);
            b1.q[0] = *(const U4*)(krow + 32 + lhi * 8);
            b1.q[1] = *(const U4*)(krow + 48 + lhi * 8);
            s[t] = __builtin_amdgcn_wmma_f32_16x16x32_bf16(
                false, qa[0].v, false, b0.v, (short)0, s[t], false, false);
            s[t] = __builtin_amdgcn_wmma_f32_16x16x32_bf16(
                false, qa[1].v, false, b1.v, (short)0, s[t], false, false);
        }

        // ---- V block staging handshake (overlapped with score math) ----
        if (threadIdx.x < 32) __builtin_amdgcn_s_wait_tensorcnt(0);
        __syncthreads();
        if ((kb + 32 < S_LEN) && threadIdx.x < 32) {
            tdm_load_tile((unsigned int)(size_t)&vtile[buf ^ 1][0],
                          vbase + (size_t)(kb + 32) * rs,
                          64u, 32u, (unsigned int)rs, 64u);
        }

        // ---- online softmax update ----
        float mnew[8], psum[8];
#pragma unroll
        for (int v = 0; v < 8; ++v) {
            float s0 = s[0][v] * scale;
            float s1 = s[1][v] * scale;
            float mx = fmaxf(s0, s1);
#pragma unroll
            for (int msk = 1; msk < 16; msk <<= 1)
                mx = fmaxf(mx, __shfl_xor(mx, msk, 32));
            mnew[v] = fmaxf(m_i[v], mx);
            float p0 = __expf(s0 - mnew[v]);
            float p1 = __expf(s1 - mnew[v]);
            s[0][v] = p0;
            s[1][v] = p1;
            float ps = p0 + p1;
#pragma unroll
            for (int msk = 1; msk < 16; msk <<= 1)
                ps += __shfl_xor(ps, msk, 32);
            psum[v] = ps;
        }
#pragma unroll
        for (int v = 0; v < 8; ++v) {
            float alpha = __expf(m_i[v] - mnew[v]);
            l_i[v] = l_i[v] * alpha + psum[v];
            m_i[v] = mnew[v];
#pragma unroll
            for (int t = 0; t < 4; ++t) o[t][v] *= alpha;
        }

        // ---- P (C-layout) -> LDS -> A-frag layout ----
        unsigned short* pw = plds[wave];
#pragma unroll
        for (int t = 0; t < 2; ++t)
#pragma unroll
            for (int v = 0; v < 8; ++v)
                pw[(v + 8 * lhi) * 32 + t * 16 + lrow] = f32_to_bf16(s[t][v]);

        BF16Frag pa;
        pa.q[0] = *(const U4*)(pw + lrow * 32 +      lhi * 8);
        pa.q[1] = *(const U4*)(pw + lrow * 32 + 16 + lhi * 8);

        // ---- O += P * V, V B-frags from LDS tile [key][dim] ----
        const unsigned short* vsm = vtile[buf];
#pragma unroll
        for (int t = 0; t < 4; ++t) {
            BF16Frag vb;
            const int n = t * 16 + lrow;
#pragma unroll
            for (int j = 0; j < 8; ++j) {
                vb.h[j]     = vsm[(     lhi * 8 + j) * 64 + n];
                vb.h[8 + j] = vsm[(16 + lhi * 8 + j) * 64 + n];
            }
            o[t] = __builtin_amdgcn_wmma_f32_16x16x32_bf16(
                false, pa.v, false, vb.v, (short)0, o[t], false, false);
        }
        __syncthreads();                  // all waves done with vtile[buf]
    }

    // ---- finalize: O /= l, write [B,S,DM] bf16 (heads re-interleaved) ----
#pragma unroll
    for (int t = 0; t < 4; ++t) {
#pragma unroll
        for (int v = 0; v < 8; ++v) {
            float val = o[t][v] / l_i[v];
            const int m = q0 + v + 8 * lhi;
            const int n = h * HD + t * 16 + lrow;
            vout[((size_t)b * S_LEN + m) * DM + n] = f32_to_bf16(val);
        }
    }
}

// ---------------------------------------------------------------------------
// Launch
// ---------------------------------------------------------------------------
extern "C" void kernel_launch(void* const* d_in, const int* in_sizes, int n_in,
                              void* d_out, int out_size, void* d_ws, size_t ws_size,
                              hipStream_t stream) {
    const float* x     = (const float*)d_in[0];   // [B,S,DM]
    const float* W_qkv = (const float*)d_in[1];   // [3*DM, DM]
    const float* b_qkv = (const float*)d_in[2];   // [3*DM]
    const float* W_out = (const float*)d_in[3];   // [DM, DM]
    const float* b_out = (const float*)d_in[4];   // [DM]

    char* ws = (char*)d_ws;
    size_t off = 0;
    auto alloc = [&](size_t bytes) -> void* {
        void* p = ws + off;
        off += (bytes + 255) & ~(size_t)255;
        return p;
    };
    unsigned short* xb    = (unsigned short*)alloc((size_t)MROWS * DM * 2);    // 16 MB
    unsigned short* wqkvb = (unsigned short*)alloc((size_t)QKV_N * DM * 2);    //  6 MB
    unsigned short* woutb = (unsigned short*)alloc((size_t)DM * DM * 2);       //  2 MB
    unsigned short* qkvb  = (unsigned short*)alloc((size_t)MROWS * QKV_N * 2); // 48 MB
    unsigned short* vbuf  = (unsigned short*)alloc((size_t)MROWS * DM * 2);    // 16 MB

    // 1) fp32 -> bf16 conversions
    {
        int n;
        n = MROWS * DM;
        f2bf_kernel<<<(n + 255) / 256, 256, 0, stream>>>(x, xb, n);
        n = QKV_N * DM;
        f2bf_kernel<<<(n + 255) / 256, 256, 0, stream>>>(W_qkv, wqkvb, n);
        n = DM * DM;
        f2bf_kernel<<<(n + 255) / 256, 256, 0, stream>>>(W_out, woutb, n);
    }

    // 2) fused QKV projection: [8192 x 1024] x [3072 x 1024]^T + b -> bf16
    gemm_bf16_kernel<<<dim3(QKV_N / 64, MROWS / 128), 256, 0, stream>>>(
        xb, wqkvb, b_qkv, qkvb, nullptr, QKV_N, DM);

    // 3) flash attention per (b,h), 16-query tiles
    attn_kernel<<<dim3(S_LEN / 16 / 4, B_SZ * NH), 128, 0, stream>>>(qkvb, vbuf);

    // 4) output projection: [8192 x 1024] x [1024 x 1024]^T + b -> fp32 out
    gemm_bf16_kernel<<<dim3(DM / 64, MROWS / 128), 256, 0, stream>>>(
        vbuf, woutb, b_out, nullptr, (float*)d_out, DM, DM);
}
